// TransformerModel_1443109012058
// MI455X (gfx1250) — matrix operand order repaired and checked by reading, not compiled, not run on hardware
//
#include <hip/hip_runtime.h>
#include <hip/hip_fp16.h>
#include <math.h>

typedef __attribute__((ext_vector_type(16))) _Float16 v16h;
typedef __attribute__((ext_vector_type(8)))  float    v8f;

#define WPB 4
#define ATTN_SCALE 0.25f   // 1/sqrt(16)

#define WMMA_F16(a, b, cin) \
  __builtin_amdgcn_wmma_f32_16x16x32_f16(false, (a), false, (b), (short)0, (cin), false, false)

// A-matrix fragment (16x32 f16): lane holds row M = lane%16.
// K mapping: K0 = half*16 + 2j (contiguous, matches B fragments)
__device__ __forceinline__ v16h frag_A(const _Float16* src, int ld, int rowBase,
                                       int kValid, int lane) {
  const int M = lane & 15, h = lane >> 4;
  const _Float16* row = src + (rowBase + M) * ld;
  v16h f;
#pragma unroll
  for (int j = 0; j < 8; ++j) {
    const int k0 = 16 * h + 2 * j;
    f[2 * j]     = (k0 < kValid)     ? row[k0]     : (_Float16)0.f;
    f[2 * j + 1] = (k0 + 1 < kValid) ? row[k0 + 1] : (_Float16)0.f;
  }
  return f;
}

// B-matrix fragment (32x16 f16, K x N) built from B^T stored row-major (srcT[n][k]).
// Contiguous K order (same as frag_A): lanes 0-15 hold K=0..15 (2/VGPR), lanes 16-31 hold K=16..31.
__device__ __forceinline__ v16h frag_B(const _Float16* srcT, int ld, int rowBase,
                                       int kValid, int lane) {
  const int N = lane & 15, h = lane >> 4;
  const _Float16* row = srcT + (rowBase + N) * ld;
  v16h f;
#pragma unroll
  for (int j = 0; j < 8; ++j) {
    const int k0 = 16 * h + 2 * j;
    f[2 * j]     = (k0 < kValid)     ? row[k0]     : (_Float16)0.f;
    f[2 * j + 1] = (k0 + 1 < kValid) ? row[k0 + 1] : (_Float16)0.f;
  }
  return f;
}

// Weight B-fragment straight from global f32 16x16 row-major W (B^T == W for '...sd,ed->...se').
__device__ __forceinline__ v16h frag_B_w16(const float* __restrict__ W, int lane) {
  const int N = lane & 15, h = lane >> 4;
  v16h f;
#pragma unroll
  for (int j = 0; j < 8; ++j) {
    const int k0 = 16 * h + 2 * j;
    if (k0 < 16) {
      f[2 * j]     = (_Float16)W[N * 16 + k0];
      f[2 * j + 1] = (_Float16)W[N * 16 + k0 + 1];
    } else {
      f[2 * j] = (_Float16)0.f; f[2 * j + 1] = (_Float16)0.f;
    }
  }
  return f;
}

// D layout: VGPR r holds (M = 8*half + r, N = lane%16)
__device__ __forceinline__ void store_D(const v8f& c, _Float16* dst, int ld,
                                        int rowBase, int lane) {
  const int N = lane & 15, h = lane >> 4;
#pragma unroll
  for (int r = 0; r < 8; ++r)
    dst[(rowBase + 8 * h + r) * ld + N] = (_Float16)c[r];
}

__device__ __forceinline__ void store_D_T(const v8f& c, _Float16* dst, int ld,
                                          int colBase, int lane) {
  const int N = lane & 15, h = lane >> 4;
#pragma unroll
  for (int r = 0; r < 8; ++r)
    dst[N * ld + colBase + 8 * h + r] = (_Float16)c[r];
}

__device__ __forceinline__ void store_D_f32(const v8f& c, float* dst, int ld,
                                            int rowBase, int colBase, int lane) {
  const int N = lane & 15, h = lane >> 4;
#pragma unroll
  for (int r = 0; r < 8; ++r)
    dst[(rowBase + 8 * h + r) * ld + colBase + N] = c[r];
}

__global__ __launch_bounds__(32 * WPB)
void encoder_sim_kernel(const int* __restrict__ tokens,
                        const float* __restrict__ emb,
                        const float* __restrict__ wq, const float* __restrict__ wk,
                        const float* __restrict__ wv, const float* __restrict__ w1,
                        const float* __restrict__ b1, const float* __restrict__ w2,
                        const float* __restrict__ b2,
                        float* __restrict__ out, int npairs) {
  __shared__ _Float16 s_x [WPB][32 * 16];   // x, then reused for y, by stage
  __shared__ _Float16 s_q [WPB][32 * 16];   // q, then reused for h
  __shared__ _Float16 s_k [WPB][32 * 16];
  __shared__ _Float16 s_vT[WPB][16 * 32];   // v transposed: [e][t]
  __shared__ float    s_sc[WPB][32 * 32];
  __shared__ _Float16 s_at[WPB][32 * 32];

  const int wave = threadIdx.x >> 5;
  const int lane = threadIdx.x & 31;
  const int pair = blockIdx.x * WPB + wave;
  if (pair >= npairs) return;            // wave-uniform: EXEC stays all-ones

  _Float16* xb  = s_x [wave];
  _Float16* qb  = s_q [wave];
  _Float16* kb  = s_k [wave];
  _Float16* vTb = s_vT[wave];
  float*    scb = s_sc[wave];
  _Float16* atb = s_at[wave];

  const v16h fwq = frag_B_w16(wq, lane);
  const v16h fwk = frag_B_w16(wk, lane);
  const v16h fwv = frag_B_w16(wv, lane);
  const v16h fw1 = frag_B_w16(w1, lane);
  const v16h fw2 = frag_B_w16(w2, lane);
  const float b1v = b1[lane & 15];
  const float b2v = b2[lane & 15];

  float vcomp[2];

  for (int c = 0; c < 2; ++c) {
    // ---- embedding gather: one token row (64B) per lane
    {
      const int tok = tokens[(pair * 2 + c) * 32 + lane];
      const float4* e4 = reinterpret_cast<const float4*>(emb + (size_t)tok * 16);
#pragma unroll
      for (int i = 0; i < 4; ++i) {
        float4 v = e4[i];
        xb[lane * 16 + 4 * i + 0] = (_Float16)v.x;
        xb[lane * 16 + 4 * i + 1] = (_Float16)v.y;
        xb[lane * 16 + 4 * i + 2] = (_Float16)v.z;
        xb[lane * 16 + 4 * i + 3] = (_Float16)v.w;
      }
    }
    // ---- q,k,v projections (K=16 padded to 32)
#pragma unroll
    for (int m = 0; m < 2; ++m) {
      v16h ax = frag_A(xb, 16, 16 * m, 16, lane);
      v8f z = {};
      v8f cq = WMMA_F16(ax, fwq, z); store_D  (cq, qb,  16, 16 * m, lane);
      v8f ck = WMMA_F16(ax, fwk, z); store_D  (ck, kb,  16, 16 * m, lane);
      v8f cv = WMMA_F16(ax, fwv, z); store_D_T(cv, vTb, 32, 16 * m, lane);
    }
    // ---- scores = q @ k^T  (4 output tiles, K=16)
#pragma unroll
    for (int m = 0; m < 2; ++m) {
      v16h aq = frag_A(qb, 16, 16 * m, 16, lane);
#pragma unroll
      for (int n = 0; n < 2; ++n) {
        v16h bk = frag_B(kb, 16, 16 * n, 16, lane);
        v8f z = {};
        v8f cs = WMMA_F16(aq, bk, z);
        store_D_f32(cs, scb, 32, 16 * m, 16 * n, lane);
      }
    }
    // ---- softmax: one 32-wide row per lane (f32 in LDS)
    {
      float mx = -3.0e38f;
#pragma unroll
      for (int t = 0; t < 32; ++t) mx = fmaxf(mx, scb[lane * 32 + t] * ATTN_SCALE);
      float sum = 0.f;
#pragma unroll
      for (int t = 0; t < 32; ++t) {
        float e = __expf(scb[lane * 32 + t] * ATTN_SCALE - mx);
        sum += e;
        atb[lane * 32 + t] = (_Float16)e;
      }
      const float inv = 1.0f / sum;
#pragma unroll
      for (int t = 0; t < 32; ++t)
        atb[lane * 32 + t] = (_Float16)((float)atb[lane * 32 + t] * inv);
    }
    // ---- y = attn @ v  (full K=32); reuse xb as y buffer
    {
      v16h bv = frag_B(vTb, 32, 0, 32, lane);
#pragma unroll
      for (int m = 0; m < 2; ++m) {
        v16h aa = frag_A(atb, 32, 16 * m, 32, lane);
        v8f z = {};
        v8f cy = WMMA_F16(aa, bv, z);
        store_D(cy, xb, 16, 16 * m, lane);
      }
    }
    // ---- h = relu(y @ w1^T + b1); reuse qb as h buffer
#pragma unroll
    for (int m = 0; m < 2; ++m) {
      v16h ay = frag_A(xb, 16, 16 * m, 16, lane);
      v8f z = {};
      v8f ch = WMMA_F16(ay, fw1, z);
#pragma unroll
      for (int r = 0; r < 8; ++r) ch[r] = fmaxf(ch[r] + b1v, 0.f);
      store_D(ch, qb, 16, 16 * m, lane);
    }
    // ---- out = h @ w2^T + b2, fused column-mean (mean over 32 rows)
    float colp = 0.f;
#pragma unroll
    for (int m = 0; m < 2; ++m) {
      v16h ah = frag_A(qb, 16, 16 * m, 16, lane);
      v8f z = {};
      v8f co = WMMA_F16(ah, fw2, z);
#pragma unroll
      for (int r = 0; r < 8; ++r) colp += co[r];
    }
    colp += __shfl_xor(colp, 16, 32);          // combine the two M-halves
    vcomp[c] = colp * (1.0f / 32.0f) + b2v;    // vec[e], e = lane%16 (replicated in both halves)
  }

  // ---- cosine similarity across e (16-lane reduction, both halves hold full copies)
  float dp = vcomp[0] * vcomp[1];
  float n1 = vcomp[0] * vcomp[0];
  float n2 = vcomp[1] * vcomp[1];
#pragma unroll
  for (int off = 8; off >= 1; off >>= 1) {
    dp += __shfl_xor(dp, off, 32);
    n1 += __shfl_xor(n1, off, 32);
    n2 += __shfl_xor(n2, off, 32);
  }
  const float sim = dp / (sqrtf(n1) * sqrtf(n2) + 1e-8f);
  if (lane == 0) out[pair] = (sim + 1.0f) * 0.5f;
}

extern "C" void kernel_launch(void* const* d_in, const int* in_sizes, int n_in,
                              void* d_out, int out_size, void* d_ws, size_t ws_size,
                              hipStream_t stream) {
  (void)n_in; (void)out_size; (void)d_ws; (void)ws_size;
  const int*   tokens = (const int*)  d_in[0];
  const float* emb    = (const float*)d_in[1];
  const float* wq     = (const float*)d_in[2];
  const float* wk     = (const float*)d_in[3];
  const float* wv     = (const float*)d_in[4];
  const float* w1     = (const float*)d_in[5];
  const float* b1     = (const float*)d_in[6];
  const float* w2     = (const float*)d_in[7];
  const float* b2     = (const float*)d_in[8];
  float* out = (float*)d_out;

  const int npairs = in_sizes[0] / 64;   // tokens shape (B, 2, 32)
  const int blocks = (npairs + WPB - 1) / WPB;
  encoder_sim_kernel<<<blocks, 32 * WPB, 0, stream>>>(
      tokens, emb, wq, wk, wv, w1, b1, w2, b2, out, npairs);
}
